// MultiHeadAttentionGPTOSS_10307921510602
// MI455X (gfx1250) — matrix operand order repaired and checked by reading, not compiled, not run on hardware
//
#include <hip/hip_runtime.h>
#include <hip/hip_bf16.h>
#include <math.h>

typedef unsigned short u16;
typedef unsigned int   u32;
typedef __attribute__((ext_vector_type(16))) __bf16 v16bf;
typedef __attribute__((ext_vector_type(8)))  float  v8f;
typedef __attribute__((ext_vector_type(4)))  int    int4v;

union Frag16 {
  u32   u[8];
  u16   s[16];
  v16bf v;
};

__device__ __forceinline__ u16 f2bf(float f) {
  u32 u = __float_as_uint(f);
  u32 r = u + 0x7FFFu + ((u >> 16) & 1u);   // round-to-nearest-even
  return (u16)(r >> 16);
}

// --------------------------------------------------------------------------
// gfx1250 async global->LDS copy (guarded; falls back to sync copy)
// Builtin signature (from compiler diagnostic): param0 = int4 AS(1)*
// --------------------------------------------------------------------------
#if defined(__has_builtin)
#if __has_builtin(__builtin_amdgcn_global_load_async_to_lds_b128) && \
    __has_builtin(__builtin_amdgcn_s_wait_asynccnt)
#define HAVE_ASYNC_LDS 1
#endif
#endif

#ifdef HAVE_ASYNC_LDS
typedef __attribute__((address_space(1))) int4v g_int4;
typedef __attribute__((address_space(3))) int4v l_int4;
__device__ __forceinline__ void async_copy16(const void* g, void* l) {
  __builtin_amdgcn_global_load_async_to_lds_b128(
      (g_int4*)(unsigned long long)g,
      (l_int4*)(unsigned int)(unsigned long long)l, 0, 0);
}
__device__ __forceinline__ void async_wait0() {
  __builtin_amdgcn_s_wait_asynccnt(0);
}
#else
__device__ __forceinline__ void async_copy16(const void* g, void* l) {
  *(uint4*)l = *(const uint4*)g;
}
__device__ __forceinline__ void async_wait0() {}
#endif

// ---------------------------------------------------------------------------
// GEMM: C[M,N](fp32) = A[M,K] @ B[K,N](fp32) + bias[N], bf16 WMMA compute.
// A_BF16=false: A is fp32, converted to bf16 during LDS staging.
// A_BF16=true : A is bf16, async-staged raw into LDS.
// Block 256 threads = 8 waves (4x2). Block tile 128x128, k-step 32.
// M % 128 == 0, K % 32 == 0; N edge handled at 16-col granularity.
// ---------------------------------------------------------------------------
template <bool A_BF16>
__global__ __launch_bounds__(256)
void gemm_wmma(const void* __restrict__ Av, const float* __restrict__ B,
               const float* __restrict__ bias, float* __restrict__ C,
               int N, int K) {
  __shared__ __align__(16) u16 As[128 * 32];   // [row][k]
  __shared__ __align__(16) u16 Bs[128 * 32];   // [n][k]  (transposed tile)

  const int bm   = blockIdx.y * 128;
  const int bn   = blockIdx.x * 128;
  const int tid  = threadIdx.x;
  const int wave = tid >> 5;
  const int lane = tid & 31;
  const int half = lane >> 4;
  const int mr   = lane & 15;
  const int wr   = wave >> 1;      // 0..3  (32-row strip)
  const int wc   = wave & 1;       // 0..1  (64-col strip)
  const int nrem = N - bn;         // multiple of 16

  v8f acc[2][4] = {};

  for (int k0 = 0; k0 < K; k0 += 32) {
    __syncthreads();
    // ---- stage A tile 128x32 ----
    if constexpr (A_BF16) {
      const u16* A = (const u16*)Av;
      #pragma unroll
      for (int q = 0; q < 2; ++q) {
        int c0  = tid * 2 + q;          // 0..511
        int row = c0 >> 2;              // 0..127
        int kk  = (c0 & 3) << 3;        // 0,8,16,24
        async_copy16(&A[(size_t)(bm + row) * K + k0 + kk], &As[row * 32 + kk]);
      }
    } else {
      const float* A = (const float*)Av;
      if (k0 + 32 < K)
        __builtin_prefetch(&A[(size_t)(bm + (tid >> 1)) * K + k0 + 32 + (tid & 1) * 16], 0, 1);
      #pragma unroll
      for (int q = 0; q < 4; ++q) {
        int c0  = tid * 4 + q;          // 0..1023
        int row = c0 >> 3;              // 0..127
        int kk  = (c0 & 7) << 2;        // 0,4,...,28
        float4 f = *(const float4*)&A[(size_t)(bm + row) * K + k0 + kk];
        uint2 pv;
        pv.x = (u32)f2bf(f.x) | ((u32)f2bf(f.y) << 16);
        pv.y = (u32)f2bf(f.z) | ((u32)f2bf(f.w) << 16);
        *(uint2*)&As[row * 32 + kk] = pv;
      }
    }
    // ---- stage B tile 32x128 fp32 -> bf16, transposed to Bs[n][k] ----
    {
      if (k0 + 32 < K)
        __builtin_prefetch(&B[(size_t)(k0 + 32 + (tid >> 3)) * N + bn + (tid & 7) * 16], 0, 1);
      int k  = tid >> 3;                // 0..31
      int n0 = (tid & 7) << 4;          // 0,16,...,112
      u16 tmp[16];
      if (n0 < nrem) {
        #pragma unroll
        for (int q = 0; q < 4; ++q) {
          float4 f = *(const float4*)&B[(size_t)(k0 + k) * N + bn + n0 + 4 * q];
          tmp[4*q+0] = f2bf(f.x); tmp[4*q+1] = f2bf(f.y);
          tmp[4*q+2] = f2bf(f.z); tmp[4*q+3] = f2bf(f.w);
        }
      } else {
        #pragma unroll
        for (int i = 0; i < 16; ++i) tmp[i] = 0;
      }
      #pragma unroll
      for (int i = 0; i < 16; ++i) Bs[(n0 + i) * 32 + k] = tmp[i];
    }
    if constexpr (A_BF16) async_wait0();
    __syncthreads();

    // ---- fragments + WMMA ----
    Frag16 a[2];
    #pragma unroll
    for (int mt = 0; mt < 2; ++mt) {
      int row = wr * 32 + mt * 16 + mr;
      #pragma unroll
      for (int r = 0; r < 8; ++r) {
        int kb = ((r >> 2) << 4) + (half << 3) + ((r & 3) << 1);
        a[mt].u[r] = *(const u32*)&As[row * 32 + kb];
      }
    }
    #pragma unroll
    for (int nt = 0; nt < 4; ++nt) {
      Frag16 b;
      int col = wc * 64 + nt * 16 + mr;
      #pragma unroll
      for (int r = 0; r < 8; ++r)
        b.u[r] = *(const u32*)&Bs[col * 32 + ((r >> 2) << 4) + (half << 3) + ((r & 3) << 1)];
      #pragma unroll
      for (int mt = 0; mt < 2; ++mt)
        acc[mt][nt] = __builtin_amdgcn_wmma_f32_16x16x32_bf16(
            false, a[mt].v, false, b.v, (short)0, acc[mt][nt], false, false);
    }
  }

  #pragma unroll
  for (int nt = 0; nt < 4; ++nt) {
    int col = bn + wc * 64 + nt * 16 + mr;
    if (col < N) {
      float bv = bias[col];
      #pragma unroll
      for (int mt = 0; mt < 2; ++mt) {
        #pragma unroll
        for (int r = 0; r < 8; ++r) {
          int row = bm + wr * 32 + mt * 16 + r + (half << 3);
          C[(size_t)row * N + col] = acc[mt][nt][r] + bv;
        }
      }
    }
  }
}

// ---------------------------------------------------------------------------
// YaRN RoPE on q,k from fp32 qkv; writes bf16 q/k/v side buffers.
// ---------------------------------------------------------------------------
__global__ __launch_bounds__(256)
void rope_kernel(const float* __restrict__ qkv, u16* __restrict__ qb,
                 u16* __restrict__ kb, u16* __restrict__ vb) {
  const int t   = blockIdx.x;
  const int tid = threadIdx.x;
  __shared__ float cs[32], sn[32];

  if (tid < 32) {
    const float two_pi  = 6.283185307179586f;
    const float ln_base = logf(150000.0f);
    const float conc    = 0.1f * logf(32.0f) + 1.0f;
    const float low     = 32.0f * logf(1024.0f / (32.0f * two_pi)) / ln_base;
    const float high    = 32.0f * logf(1024.0f / two_pi) / ln_base;
    float i_f  = (float)tid;
    float ramp = fminf(1.0f, fmaxf(0.0f, (i_f - low) / (high - low)));
    float freq = expf(ln_base * i_f / 32.0f);
    float inv_freq = ramp / (32.0f * freq) + (1.0f - ramp) / freq;
    float ang = (float)t * inv_freq;
    cs[tid] = cosf(ang) * conc;
    sn[tid] = sinf(ang) * conc;
  }
  __syncthreads();

  const float* row = qkv + (size_t)t * 5120;
  for (int p = tid; p < 2048; p += 256) {      // q: 64 heads x 32 pairs
    int hh = p >> 5, i = p & 31;
    float x1 = row[hh * 64 + i], x2 = row[hh * 64 + i + 32];
    float c = cs[i], s = sn[i];
    qb[((size_t)t * 64 + hh) * 64 + i]      = f2bf(x1 * c - x2 * s);
    qb[((size_t)t * 64 + hh) * 64 + i + 32] = f2bf(x2 * c + x1 * s);
  }
  for (int p = tid; p < 256; p += 256) {       // k: 8 heads x 32 pairs
    int hh = p >> 5, i = p & 31;
    const float* kr = row + 4096 + hh * 64;
    float x1 = kr[i], x2 = kr[i + 32];
    float c = cs[i], s = sn[i];
    kb[((size_t)t * 8 + hh) * 64 + i]      = f2bf(x1 * c - x2 * s);
    kb[((size_t)t * 8 + hh) * 64 + i + 32] = f2bf(x2 * c + x1 * s);
  }
  for (int e = tid; e < 512; e += 256)         // v copy
    vb[(size_t)t * 512 + e] = f2bf(row[4608 + e]);
}

// ---------------------------------------------------------------------------
// Sliding-window attention with sink.
// Grid (96 q-tiles, 8 kv-heads); 256 threads = 8 waves, one q-head per wave.
// K/V 144-key window async-staged once per block. Register softmax.
// Dynamic LDS: Kl[160][64] + Vl[160][64] + Wl[8][16][160] = 80 KB
// ---------------------------------------------------------------------------
__global__ __launch_bounds__(256)
void attn_wmma(const u16* __restrict__ qb, const u16* __restrict__ kb,
               const u16* __restrict__ vb, const float* __restrict__ sinks,
               u16* __restrict__ attn_out) {
  extern __shared__ __align__(16) u16 smem[];
  u16* Kl = smem;                    // [160][64]
  u16* Vl = smem + 160 * 64;         // [160][64]
  u16* Wp = smem + 2 * 160 * 64;     // [8][16][160]

  const int hkv  = blockIdx.y;
  const int q0   = blockIdx.x * 16;
  const int tid  = threadIdx.x;
  const int wave = tid >> 5;
  const int lane = tid & 31;
  const int h    = hkv * 8 + wave;
  const int half = lane >> 4;
  const int mr   = lane & 15;

  // fixed 144-key window: keys q0-128 .. q0+15 at kl = keyg - (q0-128)
  const int start_kl = (q0 >= 128) ? 0 : (128 - q0);
  const int kt0f = (q0 >> 4) - 8;

  // zero V rows not covered by the copy (disjoint from async writes)
  for (int p = tid; p < start_kl * 8; p += 256) {
    uint4 z = {0, 0, 0, 0};
    *(uint4*)&Vl[(p >> 3) * 64 + (p & 7) * 8] = z;
  }
  for (int p = tid; p < 16 * 8; p += 256) {    // rows 144..159
    uint4 z = {0, 0, 0, 0};
    *(uint4*)&Vl[(144 + (p >> 3)) * 64 + (p & 7) * 8] = z;
  }
  for (int c = tid; c < (144 - start_kl) * 8; c += 256) {
    int kl = start_kl + (c >> 3);
    int ch = (c & 7) * 8;
    size_t gi = ((size_t)(q0 - 128 + kl) * 8 + hkv) * 64 + ch;
    async_copy16(&kb[gi], &Kl[kl * 64 + ch]);
    async_copy16(&vb[gi], &Vl[kl * 64 + ch]);
  }
  async_wait0();
  __syncthreads();

  // Q fragments (two 32-wide chunks over D=64)
  Frag16 qa[2];
  #pragma unroll
  for (int s = 0; s < 2; ++s) {
    #pragma unroll
    for (int r = 0; r < 8; ++r) {
      int d = s * 32 + ((r >> 2) << 4) + (half << 3) + ((r & 3) << 1);
      qa[s].u[r] = *(const u32*)&qb[(((size_t)(q0 + mr)) * 64 + h) * 64 + d];
    }
  }

  // S = Q @ K^T over fixed 9 key tiles (tiles with kt<0 fully masked)
  v8f st[9];
  #pragma unroll
  for (int i = 0; i < 9; ++i) {
    int kt = kt0f + i;
    v8f sacc = {};
    if (kt >= 0) {
      int keyl = i * 16 + mr;
      #pragma unroll
      for (int s = 0; s < 2; ++s) {
        Frag16 bk;
        #pragma unroll
        for (int r = 0; r < 8; ++r)
          bk.u[r] = *(const u32*)&Kl[keyl * 64 + s * 32 + ((r >> 2) << 4) + (half << 3) + ((r & 3) << 1)];
        sacc = __builtin_amdgcn_wmma_f32_16x16x32_bf16(
            false, qa[s].v, false, bk.v, (short)0, sacc, false, false);
      }
    }
    #pragma unroll
    for (int r = 0; r < 8; ++r) {
      int m = r + (half << 3);
      int q = q0 + m;
      int key = kt * 16 + mr;
      float v = sacc[r] * 0.125f;                       // 1/sqrt(64)
      if (kt < 0 || key > q || key + 128 <= q) v = -1.0e30f;
      st[i][r] = v;
    }
  }

  // register softmax: row m = r + 8*half lives across each 16-lane group
  const float sinkv = sinks[h];
  u16* Wl = Wp + wave * 16 * 160;
  #pragma unroll
  for (int r = 0; r < 8; ++r) {
    float mx = -3.0e38f;
    #pragma unroll
    for (int i = 0; i < 9; ++i) mx = fmaxf(mx, st[i][r]);
    mx = fmaxf(mx, __shfl_xor(mx, 1));
    mx = fmaxf(mx, __shfl_xor(mx, 2));
    mx = fmaxf(mx, __shfl_xor(mx, 4));
    mx = fmaxf(mx, __shfl_xor(mx, 8));
    float sum = 0.0f;
    #pragma unroll
    for (int i = 0; i < 9; ++i) {
      float e = __expf(st[i][r] - mx);
      st[i][r] = e;
      sum += e;
    }
    sum += __shfl_xor(sum, 1);
    sum += __shfl_xor(sum, 2);
    sum += __shfl_xor(sum, 4);
    sum += __shfl_xor(sum, 8);
    float rd = 1.0f / (sum + __expf(sinkv - mx));
    #pragma unroll
    for (int i = 0; i < 9; ++i)
      Wl[(r + (half << 3)) * 160 + i * 16 + mr] = f2bf(st[i][r] * rd);
  }
  for (int p = lane; p < 16 * 16; p += 32)     // zero pad prob cols 144..159
    Wl[(p >> 4) * 160 + 144 + (p & 15)] = 0;
  __syncthreads();

  // O = W @ V over 160 zero-padded keys (5 k-steps), D=64 -> 4 n-tiles
  v8f oacc[4] = {};
  #pragma unroll
  for (int ks = 0; ks < 5; ++ks) {
    Frag16 wa;
    #pragma unroll
    for (int r = 0; r < 8; ++r) {
      int kk = ks * 32 + ((r >> 2) << 4) + (half << 3) + ((r & 3) << 1);
      wa.u[r] = *(const u32*)&Wl[mr * 160 + kk];
    }
    #pragma unroll
    for (int nt = 0; nt < 4; ++nt) {
      Frag16 bv;
      #pragma unroll
      for (int r = 0; r < 8; ++r) {
        #pragma unroll
        for (int j = 0; j < 2; ++j) {
          int kl = ks * 32 + ((r >> 2) << 4) + (half << 3) + ((r & 3) << 1) + j;
          bv.s[2 * r + j] = Vl[kl * 64 + nt * 16 + mr];
        }
      }
      oacc[nt] = __builtin_amdgcn_wmma_f32_16x16x32_bf16(
          false, wa.v, false, bv.v, (short)0, oacc[nt], false, false);
    }
  }

  #pragma unroll
  for (int nt = 0; nt < 4; ++nt) {
    #pragma unroll
    for (int r = 0; r < 8; ++r) {
      int row = q0 + r + (half << 3);
      int col = h * 64 + nt * 16 + mr;
      attn_out[(size_t)row * 4096 + col] = f2bf(oacc[nt][r]);
    }
  }
}

// ---------------------------------------------------------------------------
// Host-side launch
// ---------------------------------------------------------------------------
static inline size_t al256(size_t x) { return (x + 255) & ~(size_t)255; }

extern "C" void kernel_launch(void* const* d_in, const int* in_sizes, int n_in,
                              void* d_out, int out_size, void* d_ws, size_t ws_size,
                              hipStream_t stream) {
  const float* x     = (const float*)d_in[0];
  const float* Wqkv  = (const float*)d_in[1];
  const float* bqkv  = (const float*)d_in[2];
  const float* Wout  = (const float*)d_in[3];
  const float* bout  = (const float*)d_in[4];
  const float* sinks = (const float*)d_in[5];

  const int T = 1536, DM = 2880, QKV = 5120, HD = 4096;

  char* ws = (char*)d_ws;
  size_t off = 0;
  float* qkvf  = (float*)(ws + off); off += al256((size_t)T * QKV * 4);
  u16*   qb2   = (u16*)(ws + off);   off += al256((size_t)T * 64 * 64 * 2);
  u16*   kb2   = (u16*)(ws + off);   off += al256((size_t)T * 8 * 64 * 2);
  u16*   vb2   = (u16*)(ws + off);   off += al256((size_t)T * 8 * 64 * 2);
  u16*   attnb = (u16*)(ws + off);   off += al256((size_t)T * HD * 2);
  (void)ws_size; (void)in_sizes; (void)n_in; (void)out_size;

  // qkv = x @ W_qkv + b_qkv  (fp32 operands, conversion fused into staging)
  gemm_wmma<false><<<dim3((QKV + 127) / 128, T / 128), 256, 0, stream>>>(
      (const void*)x, Wqkv, bqkv, qkvf, QKV, DM);

  // RoPE + bf16 repack of q/k/v
  rope_kernel<<<T, 256, 0, stream>>>(qkvf, qb2, kb2, vb2);

  // sliding-window attention with sink (async K/V staging)
  const int attn_lds = (2 * 160 * 64 + 8 * 16 * 160) * (int)sizeof(u16);  // 80 KB
  attn_wmma<<<dim3(T / 16, 8), 256, attn_lds, stream>>>(qb2, kb2, vb2, sinks, attnb);

  // out = attn(bf16, async-staged) @ W_out(fp32) + b_out
  gemm_wmma<true><<<dim3((DM + 127) / 128, T / 128), 256, 0, stream>>>(
      (const void*)attnb, Wout, bout, (float*)d_out, DM, HD);
}
